// ReadoutLayer_103079215143
// MI455X (gfx1250) — compile-verified, matrix operand order checked by reading, not run on hardware
//
#include <hip/hip_runtime.h>

typedef __attribute__((ext_vector_type(16))) __bf16 v16bf;
typedef __attribute__((ext_vector_type(8)))  __bf16 v8bf;
typedef __attribute__((ext_vector_type(8)))  float  v8f;
typedef __attribute__((ext_vector_type(4)))  float  v4f;
typedef __attribute__((ext_vector_type(4)))  int    i32x4;

#define B_  32
#define T_  2048
#define I_  1024
#define H_  512
#define M_  (B_*T_)      /* 65536 rows */

#define BM 128
#define BN 128
#define BK 32
#define KTILES (I_/BK)

#define A_MIN_ 0.8187307530779818f   /* exp(-1/5)  */
#define A_MAX_ 0.9607894391523232f   /* exp(-1/25) */
#define EPS_   1e-5f

// ---------------------------------------------------------------------------
// CDNA5 async global->LDS path (guarded; falls back to sync staging)
// Builtin signature (from hipcc diagnostic): (int32x4 AS(1)*, int32x4 AS(3)*, Imm, Imm)
// ---------------------------------------------------------------------------
#if __has_builtin(__builtin_amdgcn_global_load_async_to_lds_b128)
#define USE_ASYNC 1
typedef __attribute__((address_space(1))) i32x4 g_i32x4;
typedef __attribute__((address_space(3))) i32x4 l_i32x4;
#else
#define USE_ASYNC 0
#endif

__device__ __forceinline__ void wait_async_zero() {
#if USE_ASYNC
#if __has_builtin(__builtin_amdgcn_s_wait_asynccnt)
    __builtin_amdgcn_s_wait_asynccnt(0);
#else
    asm volatile("s_wait_asynccnt 0x0" ::: "memory");
#endif
#endif
}

__device__ __forceinline__ void copy16B_to_lds(const __bf16* g, __bf16* l) {
#if USE_ASYNC
    __builtin_amdgcn_global_load_async_to_lds_b128((g_i32x4*)g, (l_i32x4*)l, 0, 0);
#else
    *reinterpret_cast<v8bf*>(l) = *reinterpret_cast<const v8bf*>(g);
#endif
}

// ---------------------------------------------------------------------------
// fp32 -> bf16 streaming conversion (8 elements per thread, exact multiples)
// ---------------------------------------------------------------------------
__global__ __launch_bounds__(256) void cvt_f32_to_bf16(
    const float* __restrict__ src, __bf16* __restrict__ dst)
{
    const size_t i = ((size_t)blockIdx.x * 256 + threadIdx.x) * 8;
    v4f a = *reinterpret_cast<const v4f*>(src + i);
    v4f b = *reinterpret_cast<const v4f*>(src + i + 4);
    v8bf o;
    #pragma unroll
    for (int j = 0; j < 4; ++j) { o[j] = (__bf16)a[j]; o[4 + j] = (__bf16)b[j]; }
    *reinterpret_cast<v8bf*>(dst + i) = o;
}

// ---------------------------------------------------------------------------
// GEMM: C[M,H] = Xb[M,I] * Wb[H,I]^T  (bf16 WMMA, fp32 accum)
// Double-buffered LDS fed by async global->LDS loads.
// ---------------------------------------------------------------------------
__global__ __launch_bounds__(256) void gemm_bf16_wmma(
    const __bf16* __restrict__ Xb, const __bf16* __restrict__ Wb,
    float* __restrict__ C)
{
    __shared__ __align__(16) __bf16 lA[2][BM * BK];
    __shared__ __align__(16) __bf16 lB[2][BN * BK];

    const int tid  = threadIdx.x;
    const int wid  = tid >> 5;     // 8 waves
    const int lane = tid & 31;
    const int m0 = blockIdx.y * BM;
    const int n0 = blockIdx.x * BN;

    // staging: 256 threads cover 128 rows x 32 bf16; each thread 16 bf16 (32B)
    const int srow = tid >> 1;            // 0..127
    const int scol = (tid & 1) * 16;      // 0 or 16

    const __bf16* gA = Xb + (size_t)(m0 + srow) * I_ + scol;
    const __bf16* gB = Wb + (size_t)(n0 + srow) * I_ + scol;

    v8f z = {};
    v8f acc[8];
    #pragma unroll
    for (int j = 0; j < 8; ++j) acc[j] = z;

    auto stage = [&](int k0, int buf) {
        copy16B_to_lds(gA + k0,     &lA[buf][srow * BK + scol]);
        copy16B_to_lds(gA + k0 + 8, &lA[buf][srow * BK + scol + 8]);
        copy16B_to_lds(gB + k0,     &lB[buf][srow * BK + scol]);
        copy16B_to_lds(gB + k0 + 8, &lB[buf][srow * BK + scol + 8]);
    };

    stage(0, 0);
    wait_async_zero();
    __syncthreads();

    // ISA 7.12.2 16-bit layout: lane<16 -> K{0..7,16..23}, lane>=16 -> K{8..15,24..31}
    const int kb = (lane >> 4) * 8;
    const int r  = lane & 15;

    for (int kt = 0; kt < KTILES; ++kt) {
        const int buf = kt & 1;
        if (kt + 1 < KTILES) stage((kt + 1) * BK, buf ^ 1);

        const __bf16* Ab = &lA[buf][0];
        const __bf16* Bb = &lB[buf][0];

        v16bf afrag;
        {
            v8bf lo = *reinterpret_cast<const v8bf*>(&Ab[(wid * 16 + r) * BK + kb]);
            v8bf hi = *reinterpret_cast<const v8bf*>(&Ab[(wid * 16 + r) * BK + kb + 16]);
            #pragma unroll
            for (int i = 0; i < 8; ++i) { afrag[i] = lo[i]; afrag[8 + i] = hi[i]; }
        }

        #pragma unroll
        for (int j = 0; j < 8; ++j) {
            v16bf bfrag;
            v8bf lo = *reinterpret_cast<const v8bf*>(&Bb[(j * 16 + r) * BK + kb]);
            v8bf hi = *reinterpret_cast<const v8bf*>(&Bb[(j * 16 + r) * BK + kb + 16]);
            #pragma unroll
            for (int i = 0; i < 8; ++i) { bfrag[i] = lo[i]; bfrag[8 + i] = hi[i]; }

            acc[j] = __builtin_amdgcn_wmma_f32_16x16x32_bf16(
                false, afrag, false, bfrag, (short)0, acc[j], false, false);
        }

        wait_async_zero();
        __syncthreads();
    }

    // C/D layout: VGPR r -> row r (lanes 0-15) / row r+8 (lanes 16-31); col = lane%16
    const int rl = lane & 15;
    const int mh = (lane >> 4) * 8;
    #pragma unroll
    for (int j = 0; j < 8; ++j) {
        #pragma unroll
        for (int rr = 0; rr < 8; ++rr) {
            C[(size_t)(m0 + wid * 16 + rr + mh) * H_ + n0 + j * 16 + rl] = acc[j][rr];
        }
    }
}

// ---------------------------------------------------------------------------
// stats init + column sum / sum-of-squares for BatchNorm
// ---------------------------------------------------------------------------
__global__ void zero_stats(float* __restrict__ stats)
{
    stats[threadIdx.x] = 0.0f;   // 1024 threads, 1024 floats
}

__global__ __launch_bounds__(512) void bn_stats(
    const float* __restrict__ C, float* __restrict__ stats)
{
    const int h    = threadIdx.x;           // 512 columns
    const int row0 = blockIdx.x * 128;      // 512 blocks * 128 rows
    float s = 0.0f, q = 0.0f;
    #pragma unroll 4
    for (int r = 0; r < 128; ++r) {
        float v = C[(size_t)(row0 + r) * H_ + h];
        s += v;
        q += v * v;
    }
    atomicAdd(&stats[h], s);
    atomicAdd(&stats[H_ + h], q);
}

// ---------------------------------------------------------------------------
// leaky-integrate scan + cumulative softmax readout
// ---------------------------------------------------------------------------
__device__ __forceinline__ float block_reduce(float v, float* sred, int tid, bool do_max)
{
    #pragma unroll
    for (int off = 16; off > 0; off >>= 1) {
        float o = __shfl_xor(v, off, 32);
        v = do_max ? fmaxf(v, o) : (v + o);
    }
    if ((tid & 31) == 0) sred[tid >> 5] = v;
    __syncthreads();
    if (tid < 16) {
        v = sred[tid];
        #pragma unroll
        for (int off = 8; off > 0; off >>= 1) {
            float o = __shfl_xor(v, off, 16);
            v = do_max ? fmaxf(v, o) : (v + o);
        }
        if (tid == 0) sred[0] = v;
    }
    __syncthreads();
    v = sred[0];
    __syncthreads();
    return v;
}

__global__ __launch_bounds__(512) void scan_softmax(
    const float* __restrict__ C, const float* __restrict__ stats,
    const float* __restrict__ alpha, const float* __restrict__ gamma,
    const float* __restrict__ beta, const float* __restrict__ ut0,
    float* __restrict__ out)
{
    __shared__ float sred[16];
    const int h = threadIdx.x;     // one thread per hidden unit
    const int b = blockIdx.x;      // one block per batch element

    const float invN  = 1.0f / (float)M_;
    const float mean  = stats[h] * invN;
    const float var   = stats[H_ + h] * invN - mean * mean;
    const float scale = gamma[h] * rsqrtf(var + EPS_);
    const float shift = beta[h] - mean * scale;

    const float a  = fminf(fmaxf(alpha[h], A_MIN_), A_MAX_);
    const float oa = 1.0f - a;

    float ut  = ut0[b * H_ + h];
    float acc = 0.0f;

    const float* row = C + (size_t)b * T_ * H_ + h;
    for (int t = 0; t < T_; ++t) {
        float wx = row[(size_t)t * H_];
        __builtin_prefetch(&row[(size_t)(t + 1) * H_], 0, 0);
        ut = a * ut + oa * (wx * scale + shift);

        float m = block_reduce(ut, sred, h, true);
        float e = __expf(ut - m);
        float s = block_reduce(e, sred, h, false);
        acc += e / s;
    }
    out[b * H_ + h] = acc;
}

// ---------------------------------------------------------------------------
extern "C" void kernel_launch(void* const* d_in, const int* in_sizes, int n_in,
                              void* d_out, int out_size, void* d_ws, size_t ws_size,
                              hipStream_t stream)
{
    const float* x     = (const float*)d_in[0];  // [B,T,I]
    const float* W     = (const float*)d_in[1];  // [H,I]
    const float* alpha = (const float*)d_in[2];  // [H]
    const float* gamma = (const float*)d_in[3];  // [H]
    const float* beta  = (const float*)d_in[4];  // [H]
    const float* ut0   = (const float*)d_in[5];  // [B,H]
    float* out = (float*)d_out;                  // [B,H]

    // workspace layout: Xb (bf16) | Wb (bf16) | C (fp32) | stats (fp32)
    __bf16* Xb  = (__bf16*)d_ws;                           // M_*I_  bf16
    __bf16* Wb  = Xb + (size_t)M_ * I_;                    // H_*I_  bf16
    float*  C   = (float*)(Wb + (size_t)H_ * I_);          // M_*H_  fp32
    float*  stats = C + (size_t)M_ * H_;                   // 2*H_   fp32

    cvt_f32_to_bf16<<<(int)(((size_t)M_ * I_) / 2048), 256, 0, stream>>>(x, Xb);
    cvt_f32_to_bf16<<<(int)(((size_t)H_ * I_) / 2048), 256, 0, stream>>>(W, Wb);

    zero_stats<<<1, 2 * H_, 0, stream>>>(stats);
    gemm_bf16_wmma<<<dim3(H_ / BN, M_ / BM), 256, 0, stream>>>(Xb, Wb, C);
    bn_stats<<<M_ / 128, H_, 0, stream>>>(C, stats);
    scan_softmax<<<B_, H_, 0, stream>>>(C, stats, alpha, gamma, beta, ut0, out);
}
